// SetConv_78408922956142
// MI455X (gfx1250) — compile-verified, matrix operand order checked by reading, not run on hardware
//
#include <hip/hip_runtime.h>
#include <hip/hip_bf16.h>

// ---------------- problem constants (from setup_inputs) ----------------
#define BATCH    4
#define NPTS     16384
#define CIN      64          // points channels
#define NP       1024        // npoint
#define NS       32          // nsample
#define R2       0.04f       // radius^2
#define K1       67          // 3 + CIN
#define K1PAD    96          // padded to 3 k-tiles of 32
#define H1       64
#define H2       128

typedef __attribute__((ext_vector_type(16))) _Float16 v16h;
typedef __attribute__((ext_vector_type(8)))  float    v8f;

union FragH {
    v16h  v;
    float4 q[2];
};

// Load a 16-element f16 WMMA A/B fragment slice for this lane.
// Per ISA 7.12.2 (16-bit A 16x32): lane L holds row m=(L&15); elements 0..7 are
// K = kBase + kHalf + 0..7 and elements 8..15 are K = kBase + 16 + kHalf + 0..7,
// where kHalf = (L>=16)*8.  Both chunks are contiguous 16B -> two b128 loads.
__device__ __forceinline__ v16h load_frag16(const _Float16* rowPtr, int kOff) {
    FragH f;
    f.q[0] = *reinterpret_cast<const float4*>(rowPtr + kOff);
    f.q[1] = *reinterpret_cast<const float4*>(rowPtr + kOff + 16);
    return f.v;
}

// ---------------- weight pre-conversion (f32 -> f16, padded) ----------------
__global__ void prep_weights(const float* __restrict__ w1, const float* __restrict__ w2,
                             _Float16* __restrict__ w1h, _Float16* __restrict__ w2h) {
    int i = blockIdx.x * blockDim.x + threadIdx.x;
    if (i < H1 * K1PAD) {
        int o = i / K1PAD, c = i % K1PAD;
        w1h[i] = (_Float16)(c < K1 ? w1[o * K1 + c] : 0.0f);
    } else {
        int j = i - H1 * K1PAD;
        if (j < H2 * H1) w2h[j] = (_Float16)w2[j];
    }
}

// ---------------- farthest point sampling ----------------
// One 1024-thread block per batch. Each thread keeps 16 points (coords + running
// min-distance) in registers; per iteration: VALU update + shuffle/LDS argmax.
__global__ void __launch_bounds__(1024) fps_kernel(const float* __restrict__ xyz,
                                                   float* __restrict__ newxyz) {
    const int b   = blockIdx.x;
    const int tid = threadIdx.x;
    const int lane = tid & 31, wid = tid >> 5;
    const float* xb = xyz + (size_t)b * NPTS * 3;

    float px[16], py[16], pz[16], dst[16];
#pragma unroll
    for (int i = 0; i < 16; ++i) {
        int p = tid * 16 + i;
        px[i] = xb[p * 3 + 0];
        py[i] = xb[p * 3 + 1];
        pz[i] = xb[p * 3 + 2];
        dst[i] = 1e10f;
    }

    __shared__ float s_c[3];
    __shared__ float s_v[32];
    __shared__ int   s_i[32];
    __shared__ int   s_w;
    if (tid == 0) { s_c[0] = px[0]; s_c[1] = py[0]; s_c[2] = pz[0]; s_w = 0; }
    __syncthreads();

    for (int s = 0; s < NP; ++s) {
        const float cx = s_c[0], cy = s_c[1], cz = s_c[2];
        if (tid == 0) {   // emit centroid s (coords of current farthest)
            newxyz[(b * NP + s) * 3 + 0] = cx;
            newxyz[(b * NP + s) * 3 + 1] = cy;
            newxyz[(b * NP + s) * 3 + 2] = cz;
        }
        float bv = -1.0f; int bi = 0;
#pragma unroll
        for (int i = 0; i < 16; ++i) {
            float dx = px[i] - cx, dy = py[i] - cy, dz = pz[i] - cz;
            float dd = dx * dx + dy * dy + dz * dz;
            float d  = fminf(dst[i], dd);
            dst[i] = d;
            if (d > bv) { bv = d; bi = tid * 16 + i; }
        }
        // wave32 argmax (ties -> lowest index, matching jnp.argmax)
        for (int off = 16; off > 0; off >>= 1) {
            float ov = __shfl_xor(bv, off, 32);
            int   oi = __shfl_xor(bi, off, 32);
            if (ov > bv || (ov == bv && oi < bi)) { bv = ov; bi = oi; }
        }
        if (lane == 0) { s_v[wid] = bv; s_i[wid] = bi; }
        __syncthreads();
        if (wid == 0) {
            float v = s_v[lane]; int ii = s_i[lane];
            for (int off = 16; off > 0; off >>= 1) {
                float ov = __shfl_xor(v, off, 32);
                int   oi = __shfl_xor(ii, off, 32);
                if (ov > v || (ov == v && oi < ii)) { v = ov; ii = oi; }
            }
            if (lane == 0) s_w = ii;
        }
        __syncthreads();
        const int w = s_w;
        if (tid == (w >> 4)) {   // owner publishes winner coords (no global load)
            int l = w & 15;
            s_c[0] = px[l]; s_c[1] = py[l]; s_c[2] = pz[l];
        }
        __syncthreads();
    }
}

// ---------------- ball query ----------------
// One wave per centroid; ballot-compacted first-32-in-index-order, early exit.
__global__ void __launch_bounds__(256) ballq_kernel(const float* __restrict__ xyz,
                                                    const float* __restrict__ newxyz,
                                                    int* __restrict__ gidx) {
    const int g    = blockIdx.x * 8 + (threadIdx.x >> 5);
    const int lane = threadIdx.x & 31;
    const int b    = g >> 10;
    const float* xb = xyz + (size_t)b * NPTS * 3;
    const float cx = newxyz[g * 3 + 0];
    const float cy = newxyz[g * 3 + 1];
    const float cz = newxyz[g * 3 + 2];

    int cnt = 0, firstP = -1;
    for (int base = 0; base < NPTS && cnt < NS; base += 32) {
        int p = base + lane;
        float dx = xb[p * 3 + 0] - cx;
        float dy = xb[p * 3 + 1] - cy;
        float dz = xb[p * 3 + 2] - cz;
        bool in = (dx * dx + dy * dy + dz * dz) <= R2;
        unsigned m = (unsigned)__ballot(in);
        if (firstP < 0 && m != 0u) firstP = base + __ffs(m) - 1;
        int pos = cnt + __popc(m & ((1u << lane) - 1u));
        if (in && pos < NS) gidx[g * NS + pos] = p;
        cnt += __popc(m);
    }
    if (cnt < NS) {                 // pad with first neighbor (reference semantics)
        if (firstP < 0) firstP = 0;
        if (lane >= cnt && lane < NS) gidx[g * NS + lane] = firstP;
    }
}

// ---------------- fused gather + MLP(67->64->128) + k-max ----------------
// One wave per group: stage 32x96 f16 feats in LDS, GEMM1 (24 wmma), ReLU back
// into the same tile, GEMM2 (32 wmma), per-channel max over the 32 samples.
__global__ void __launch_bounds__(256) mlp_kernel(const float* __restrict__ xyz,
                                                  const float* __restrict__ points,
                                                  const _Float16* __restrict__ w1h,
                                                  const float* __restrict__ b1,
                                                  const _Float16* __restrict__ w2h,
                                                  const float* __restrict__ b2,
                                                  const int* __restrict__ gidx,
                                                  const float* __restrict__ newxyz,
                                                  float* __restrict__ outp) {
    __shared__ __align__(16) _Float16 sFeat[8][NS][K1PAD];   // 48 KB

    const int wave = threadIdx.x >> 5;
    const int lane = threadIdx.x & 31;
    const int g    = blockIdx.x * 8 + wave;
    const int b    = g >> 10;
    const int nlo  = lane & 15;
    const int kHalf = (lane >> 4) * 8;     // also the D-matrix m-half offset

    // -- gather: lane = neighbor j; build feats row [xyz-center | points | 0-pad]
    {
        int p = gidx[g * NS + lane];
        const float cx = newxyz[g * 3 + 0];
        const float cy = newxyz[g * 3 + 1];
        const float cz = newxyz[g * 3 + 2];
        const float* pr = xyz + ((size_t)b * NPTS + p) * 3;
        const float* fr = points + ((size_t)b * NPTS + p) * CIN;
        _Float16* row = &sFeat[wave][lane][0];
        row[0] = (_Float16)(pr[0] - cx);
        row[1] = (_Float16)(pr[1] - cy);
        row[2] = (_Float16)(pr[2] - cz);
#pragma unroll
        for (int c = 0; c < CIN; ++c) row[3 + c] = (_Float16)fr[c];
#pragma unroll
        for (int c = K1; c < K1PAD; ++c) row[c] = (_Float16)0.0f;
    }
    // Same-wave LDS ops are in-order (DScnt) -> no barrier needed.

    // -- GEMM1: feats(32x96) @ w1^T(96x64) + b1, ReLU
    v8f acc[2][4];
#pragma unroll
    for (int mt = 0; mt < 2; ++mt)
#pragma unroll
        for (int nt = 0; nt < 4; ++nt) {
            float bv = b1[nt * 16 + nlo];
#pragma unroll
            for (int r = 0; r < 8; ++r) acc[mt][nt][r] = bv;
        }
#pragma unroll
    for (int kt = 0; kt < 3; ++kt) {
        v16h a0 = load_frag16(&sFeat[wave][0 * 16 + nlo][0], kt * 32 + kHalf);
        v16h a1 = load_frag16(&sFeat[wave][1 * 16 + nlo][0], kt * 32 + kHalf);
#pragma unroll
        for (int nt = 0; nt < 4; ++nt) {
            v16h bf = load_frag16(w1h + (nt * 16 + nlo) * K1PAD, kt * 32 + kHalf);
            acc[0][nt] = __builtin_amdgcn_wmma_f32_16x16x32_f16(
                false, a0, false, bf, (short)0, acc[0][nt], false, false);
            acc[1][nt] = __builtin_amdgcn_wmma_f32_16x16x32_f16(
                false, a1, false, bf, (short)0, acc[1][nt], false, false);
        }
    }
    // -- ReLU(h) stored back into the (fully consumed) feats tile, cols 0..63
#pragma unroll
    for (int mt = 0; mt < 2; ++mt)
#pragma unroll
        for (int nt = 0; nt < 4; ++nt)
#pragma unroll
            for (int r = 0; r < 8; ++r) {
                int m = mt * 16 + kHalf + r;
                sFeat[wave][m][nt * 16 + nlo] = (_Float16)fmaxf(acc[mt][nt][r], 0.0f);
            }

    // -- GEMM2: h(32x64) @ w2^T(64x128) + b2, then max over the 32 samples
    v16h ha[2][2];
#pragma unroll
    for (int kt = 0; kt < 2; ++kt)
#pragma unroll
        for (int mt = 0; mt < 2; ++mt)
            ha[mt][kt] = load_frag16(&sFeat[wave][mt * 16 + nlo][0], kt * 32 + kHalf);

    float* np = outp + (size_t)g * H2;
#pragma unroll
    for (int nt = 0; nt < 8; ++nt) {
        v8f a2[2];
        float bv = b2[nt * 16 + nlo];
#pragma unroll
        for (int mt = 0; mt < 2; ++mt)
#pragma unroll
            for (int r = 0; r < 8; ++r) a2[mt][r] = bv;
#pragma unroll
        for (int kt = 0; kt < 2; ++kt) {
            v16h bf = load_frag16(w2h + (nt * 16 + nlo) * H1, kt * 32 + kHalf);
            a2[0] = __builtin_amdgcn_wmma_f32_16x16x32_f16(
                false, ha[0][kt], false, bf, (short)0, a2[0], false, false);
            a2[1] = __builtin_amdgcn_wmma_f32_16x16x32_f16(
                false, ha[1][kt], false, bf, (short)0, a2[1], false, false);
        }
        // max over m: 16-way per lane, then fold the two lane-halves
        float pm = a2[0][0];
#pragma unroll
        for (int mt = 0; mt < 2; ++mt)
#pragma unroll
            for (int r = 0; r < 8; ++r) pm = fmaxf(pm, a2[mt][r]);
        float other = __shfl_xor(pm, 16, 32);
        pm = fmaxf(fmaxf(pm, other), 0.0f);   // ReLU commutes with max
        if (lane < 16) np[nt * 16 + lane] = pm;
    }
}

// ---------------- launch ----------------
extern "C" void kernel_launch(void* const* d_in, const int* in_sizes, int n_in,
                              void* d_out, int out_size, void* d_ws, size_t ws_size,
                              hipStream_t stream) {
    const float* xyz    = (const float*)d_in[0];
    const float* points = (const float*)d_in[1];
    const float* w1     = (const float*)d_in[2];
    const float* b1     = (const float*)d_in[3];
    const float* w2     = (const float*)d_in[4];
    const float* b2     = (const float*)d_in[5];

    float* newxyz = (float*)d_out;                        // (B, NP, 3)
    float* newpts = (float*)d_out + BATCH * NP * 3;       // (B, NP, 128)

    // workspace: group indices (512 KB) + f16 weights (28 KB)
    int*      gidx = (int*)d_ws;
    _Float16* w1h  = (_Float16*)(gidx + BATCH * NP * NS);
    _Float16* w2h  = w1h + H1 * K1PAD;

    prep_weights<<<(H1 * K1PAD + H2 * H1 + 255) / 256, 256, 0, stream>>>(w1, w2, w1h, w2h);
    fps_kernel<<<BATCH, 1024, 0, stream>>>(xyz, newxyz);
    ballq_kernel<<<(BATCH * NP) / 8, 256, 0, stream>>>(xyz, newxyz, gidx);
    mlp_kernel<<<(BATCH * NP) / 8, 256, 0, stream>>>(xyz, points, w1h, b1, w2h, b2,
                                                     gidx, newxyz, newpts);
}
